// MultiHeadAttentionLayer_1271310320009
// MI455X (gfx1250) — matrix operand order repaired and checked by reading, not compiled, not run on hardware
//
#include <hip/hip_runtime.h>
#include <hip/hip_bf16.h>
#include <math.h>

typedef __attribute__((ext_vector_type(16))) _Float16 v16h;
typedef __attribute__((ext_vector_type(8)))  float    v8f;

#define BB 2
#define SS 2048
#define DD 512
#define HH 8
#define HD 64

static __device__ __forceinline__ v8f wmma_f16(v16h a, v16h b, v8f c) {
  // 8 args: (neg_a, A, neg_b, B, c_mod, C, reuse_a, reuse_b)
  return __builtin_amdgcn_wmma_f32_16x16x32_f16(false, a, false, b, (short)0, c, false, false);
}

// A fragment (16x32, f16) from a row-major [M][K] tile, leading dim ld.
// Lane m = lane%16 holds K = {hf*8..hf*8+7, 16+hf*8..16+hf*8+7}, hf = lane/16.
template <typename T>
static __device__ __forceinline__ v16h load_a(const T* p, int ld) {
  const int lane = threadIdx.x & 31;
  const int m = lane & 15, hf = lane >> 4;
  const T* row = p + (size_t)m * ld;
  v16h a;
#pragma unroll
  for (int v = 0; v < 8; ++v) {
    const int k0 = (v >> 2) * 16 + hf * 8 + (v & 3) * 2;
    a[2 * v]     = (_Float16)row[k0];
    a[2 * v + 1] = (_Float16)row[k0 + 1];
  }
  return a;
}

// B fragment (32x16, f16) sourced from a TRANSPOSED image [N][K], leading dim ld.
// Lane column n = lane%16; VGPR v holds K = (v/4)*16 + (lane/16)*8 + 2*(v%4), +1 -> contiguous pairs.
static __device__ __forceinline__ v16h load_bt(const _Float16* p, int ld) {
  const int lane = threadIdx.x & 31;
  const int n = lane & 15, hf = lane >> 4;
  const _Float16* row = p + (size_t)n * ld;
  v16h b;
#pragma unroll
  for (int v = 0; v < 8; ++v) {
    const int k0 = (v >> 2) * 16 + hf * 8 + (v & 3) * 2;
    b[2 * v]     = row[k0];
    b[2 * v + 1] = row[k0 + 1];
  }
  return b;
}

// ---------------------------------------------------------------------------
// Kernel 0: transpose + f16-convert the four weight matrices: WT[o][i] = W[i][o]
// ---------------------------------------------------------------------------
__global__ void prep_weights(const float* Wq, const float* Wk, const float* Wv, const float* Wo,
                             _Float16* WqT, _Float16* WkT, _Float16* WvT, _Float16* WoT) {
  const int w = blockIdx.y;
  const float* src = (w == 0) ? Wq : (w == 1) ? Wk : (w == 2) ? Wv : Wo;
  _Float16* dst    = (w == 0) ? WqT : (w == 1) ? WkT : (w == 2) ? WvT : WoT;
  const int idx = blockIdx.x * blockDim.x + threadIdx.x;
  if (idx < DD * DD) {
    const int o = idx & (DD - 1);
    const int i = idx >> 9;
    dst[(size_t)o * DD + i] = (_Float16)src[(size_t)i * DD + o];
  }
}

// ---------------------------------------------------------------------------
// Kernel 1: fused QKV projection, WMMA GEMM, register-blocked: each wave owns
// a 16x64 output tile -> the (expensive, f32-converted) A fragment is loaded
// once per K-step and feeds 4 wmma. Q gets 1/sqrt(HD)*log2(e) folded in.
// V is written transposed per head: VhT[b][h][hd][s].
// ---------------------------------------------------------------------------
__global__ void __launch_bounds__(128) qkv_proj(
    const float* xq, const float* xk, const float* xv,
    const _Float16* WqT, const _Float16* WkT, const _Float16* WvT,
    const float* bq, const float* bk, const float* bv,
    _Float16* Qh, _Float16* Kh, _Float16* VhT) {
  const int w = threadIdx.x >> 5;
  const int lane = threadIdx.x & 31;
  const int n = lane & 15, hf = lane >> 4;
  const int p  = blockIdx.z;        // 0=Q 1=K 2=V
  const int b  = blockIdx.y;
  const int og = blockIdx.x & 1;    // D/256 groups
  const int st = blockIdx.x >> 1;   // s tile
  const int s0 = st * 16;
  const int o0 = og * 256 + w * 64; // this wave: columns o0..o0+63

  const float*    x    = (p == 0) ? xq  : (p == 1) ? xk  : xv;
  const _Float16* WT   = (p == 0) ? WqT : (p == 1) ? WkT : WvT;
  const float*    bias = (p == 0) ? bq  : (p == 1) ? bk  : bv;

  const float* xbase = x + ((size_t)b * SS + s0) * DD;
  v8f c[4];
  const v8f vzero = {};
#pragma unroll
  for (int t = 0; t < 4; ++t) c[t] = vzero;

  for (int kk = 0; kk < DD; kk += 32) {
    const v16h a = load_a(xbase + kk, DD);   // reused by 4 wmma
#pragma unroll
    for (int t = 0; t < 4; ++t) {
      v16h bf = load_bt(WT + (size_t)(o0 + t * 16) * DD + kk, DD);
      c[t] = wmma_f16(a, bf, c[t]);
    }
  }

  const float qscale = 0.125f * 1.44269504088896f;  // 1/sqrt(64) * log2(e)
#pragma unroll
  for (int t = 0; t < 4; ++t) {
    const int o = o0 + t * 16 + n;
    const float bs = bias[o];
    const int h = o >> 6, hd = o & 63;
#pragma unroll
    for (int r = 0; r < 8; ++r) {
      const int s = s0 + r + hf * 8;
      const float val = c[t][r] + bs;
      if (p == 0) {
        Qh[(((size_t)b * HH + h) * SS + s) * HD + hd] = (_Float16)(val * qscale);
      } else if (p == 1) {
        Kh[(((size_t)b * HH + h) * SS + s) * HD + hd] = (_Float16)val;
      } else {
        VhT[(((size_t)b * HH + h) * HD + hd) * SS + s] = (_Float16)val;
      }
    }
  }
}

// ---------------------------------------------------------------------------
// Kernel 2: flash attention, register-blocked: one wave per (b, h, 32-row q
// block). Each K/V B-fragment loaded from memory feeds 2 wmma (one per
// q-tile): 8 QK^T + 8 P.V wmma per 32-key block. exp2-domain online softmax,
// P staged via per-wave LDS (C-layout -> A-layout).
// ---------------------------------------------------------------------------
__global__ void __launch_bounds__(128) attention(const _Float16* Qh, const _Float16* Kh,
                                                 const _Float16* VhT, _Float16* X) {
  __shared__ _Float16 Pbuf[4][32 * 32];  // per wave: 32 q rows x 32 keys
  const int w = threadIdx.x >> 5;
  const int lane = threadIdx.x & 31;
  const int n = lane & 15, hf = lane >> 4;
  int gw = blockIdx.x * 4 + w;
  const int qt = gw & 63;   // S/32 q-blocks
  gw >>= 6;
  const int h = gw & 7;
  const int b = gw >> 3;
  const int q0 = qt * 32;

  const _Float16* qbase = Qh + (((size_t)b * HH + h) * SS + q0) * HD;
  v16h qa[2][2];  // [qtile][kk]
#pragma unroll
  for (int q = 0; q < 2; ++q) {
    qa[q][0] = load_a(qbase + (size_t)(q * 16) * HD, HD);       // d 0..31
    qa[q][1] = load_a(qbase + (size_t)(q * 16) * HD + 32, HD);  // d 32..63
  }

  const _Float16* kh = Kh  + (((size_t)b * HH + h) * SS) * HD;
  const _Float16* vt = VhT + (((size_t)b * HH + h) * HD) * SS;

  float mrow[2][8], lrow[2][8];
  v8f oacc[2][4];
  const v8f vzero = {};
#pragma unroll
  for (int q = 0; q < 2; ++q) {
#pragma unroll
    for (int r = 0; r < 8; ++r) { mrow[q][r] = -1e30f; lrow[q][r] = 0.0f; }
#pragma unroll
    for (int t = 0; t < 4; ++t) oacc[q][t] = vzero;
  }

  _Float16* P = &Pbuf[w][0];

  for (int j = 0; j < SS / 32; ++j) {
    const _Float16* kb = kh + (size_t)(j * 32) * HD;
    __builtin_prefetch(kb + 32 * HD, 0, 0);  // global_prefetch next K block

    // K block B fragments (shared by both q-tiles).
    v16h kbf[2][2];  // [kt][kk]
#pragma unroll
    for (int kt = 0; kt < 2; ++kt) {
      kbf[kt][0] = load_bt(kb + (size_t)(kt * 16) * HD, HD);       // d 0..31
      kbf[kt][1] = load_bt(kb + (size_t)(kt * 16) * HD + 32, HD);  // d 32..63
    }

    v8f e[2][2];  // [qtile][kt]
#pragma unroll
    for (int q = 0; q < 2; ++q) {
#pragma unroll
      for (int kt = 0; kt < 2; ++kt) {
        v8f z = vzero;
        z = wmma_f16(qa[q][0], kbf[kt][0], z);
        e[q][kt] = wmma_f16(qa[q][1], kbf[kt][1], z);
      }
    }

    // Online softmax per q-tile (exp2 domain; scale folded into Q).
#pragma unroll
    for (int q = 0; q < 2; ++q) {
#pragma unroll
      for (int r = 0; r < 8; ++r) {
        float t = fmaxf(e[q][0][r], e[q][1][r]);
        t = fmaxf(t, __shfl_xor(t, 1, 32));
        t = fmaxf(t, __shfl_xor(t, 2, 32));
        t = fmaxf(t, __shfl_xor(t, 4, 32));
        t = fmaxf(t, __shfl_xor(t, 8, 32));  // stays within 16-lane half group
        const float mnew  = fmaxf(mrow[q][r], t);
        const float alpha = exp2f(mrow[q][r] - mnew);
        mrow[q][r] = mnew;
        const float p0 = exp2f(e[q][0][r] - mnew);
        const float p1 = exp2f(e[q][1][r] - mnew);
        lrow[q][r] = lrow[q][r] * alpha + p0 + p1;  // per-lane partial
        const int row = q * 16 + r + hf * 8;
        P[row * 32 + n]      = (_Float16)p0;
        P[row * 32 + 16 + n] = (_Float16)p1;
#pragma unroll
        for (int t4 = 0; t4 < 4; ++t4) oacc[q][t4][r] *= alpha;
      }
    }
    asm volatile("s_wait_dscnt 0" ::: "memory");  // LDS RAW before A-frag read

    v16h pa[2];
#pragma unroll
    for (int q = 0; q < 2; ++q) pa[q] = load_a(P + (size_t)(q * 16) * 32, 32);

#pragma unroll
    for (int t = 0; t < 4; ++t) {
      // VhT is [d][s] == [N][K] image for P.V; K = 32 keys of this block.
      v16h vb = load_bt(vt + (size_t)(t * 16) * SS + j * 32, SS);
#pragma unroll
      for (int q = 0; q < 2; ++q) oacc[q][t] = wmma_f16(pa[q], vb, oacc[q][t]);
    }
  }

  // Finish: reduce l across the 16 columns, normalize, store f16 X[b][s][h*64+d].
#pragma unroll
  for (int q = 0; q < 2; ++q) {
#pragma unroll
    for (int r = 0; r < 8; ++r) {
      float l = lrow[q][r];
      l += __shfl_xor(l, 1, 32);
      l += __shfl_xor(l, 2, 32);
      l += __shfl_xor(l, 4, 32);
      l += __shfl_xor(l, 8, 32);
      lrow[q][r] = 1.0f / l;
    }
#pragma unroll
    for (int t = 0; t < 4; ++t) {
#pragma unroll
      for (int r = 0; r < 8; ++r) {
        const int row = q * 16 + r + hf * 8;
        X[((size_t)b * SS + q0 + row) * DD + h * HD + t * 16 + n] =
            (_Float16)(oacc[q][t][r] * lrow[q][r]);
      }
    }
  }
}

// ---------------------------------------------------------------------------
// Kernel 3: output projection, WMMA GEMM, register-blocked 16x64 per wave,
// f32 output + bias.
// ---------------------------------------------------------------------------
__global__ void __launch_bounds__(128) out_proj(const _Float16* X, const _Float16* WoT,
                                                const float* bo, float* out) {
  const int w = threadIdx.x >> 5;
  const int lane = threadIdx.x & 31;
  const int n = lane & 15, hf = lane >> 4;
  const int b  = blockIdx.y;
  const int og = blockIdx.x & 1;
  const int st = blockIdx.x >> 1;
  const int s0 = st * 16;
  const int o0 = og * 256 + w * 64;

  const _Float16* xbase = X + ((size_t)b * SS + s0) * DD;
  v8f c[4];
  const v8f vzero = {};
#pragma unroll
  for (int t = 0; t < 4; ++t) c[t] = vzero;

  for (int kk = 0; kk < DD; kk += 32) {
    const v16h a = load_a(xbase + kk, DD);
#pragma unroll
    for (int t = 0; t < 4; ++t) {
      v16h bf = load_bt(WoT + (size_t)(o0 + t * 16) * DD + kk, DD);
      c[t] = wmma_f16(a, bf, c[t]);
    }
  }
#pragma unroll
  for (int t = 0; t < 4; ++t) {
    const float bs = bo[o0 + t * 16 + n];
#pragma unroll
    for (int r = 0; r < 8; ++r) {
      const int row = r + hf * 8;
      out[((size_t)b * SS + s0 + row) * DD + o0 + t * 16 + n] = c[t][r] + bs;
    }
  }
}

// ---------------------------------------------------------------------------
extern "C" void kernel_launch(void* const* d_in, const int* in_sizes, int n_in,
                              void* d_out, int out_size, void* d_ws, size_t ws_size,
                              hipStream_t stream) {
  (void)in_sizes; (void)n_in; (void)out_size; (void)ws_size;
  const float* query = (const float*)d_in[0];
  const float* key   = (const float*)d_in[1];
  const float* value = (const float*)d_in[2];
  const float* Wq = (const float*)d_in[3];
  const float* bq = (const float*)d_in[4];
  const float* Wk = (const float*)d_in[5];
  const float* bk = (const float*)d_in[6];
  const float* Wv = (const float*)d_in[7];
  const float* bv = (const float*)d_in[8];
  const float* Wo = (const float*)d_in[9];
  const float* bo = (const float*)d_in[10];
  float* out = (float*)d_out;

  char* ws = (char*)d_ws;
  size_t off = 0;
  auto alloc = [&](size_t bytes) -> char* {
    char* p = ws + off;
    off += (bytes + 255) & ~(size_t)255;
    return p;
  };
  _Float16* WqT = (_Float16*)alloc((size_t)DD * DD * 2);
  _Float16* WkT = (_Float16*)alloc((size_t)DD * DD * 2);
  _Float16* WvT = (_Float16*)alloc((size_t)DD * DD * 2);
  _Float16* WoT = (_Float16*)alloc((size_t)DD * DD * 2);
  _Float16* Qh  = (_Float16*)alloc((size_t)BB * HH * SS * HD * 2);
  _Float16* Kh  = (_Float16*)alloc((size_t)BB * HH * SS * HD * 2);
  _Float16* VhT = (_Float16*)alloc((size_t)BB * HH * SS * HD * 2);
  _Float16* X   = (_Float16*)alloc((size_t)BB * SS * DD * 2);

  prep_weights<<<dim3((DD * DD + 255) / 256, 4), 256, 0, stream>>>(
      Wq, Wk, Wv, Wo, WqT, WkT, WvT, WoT);
  qkv_proj<<<dim3((SS / 16) * (DD / 256), BB, 3), 128, 0, stream>>>(
      query, key, value, WqT, WkT, WvT, bq, bk, bv, Qh, Kh, VhT);
  attention<<<dim3(BB * HH * (SS / 32) / 4), 128, 0, stream>>>(Qh, Kh, VhT, X);
  out_proj<<<dim3((SS / 16) * (DD / 256), BB), 128, 0, stream>>>(X, WoT, bo, out);
}
